// LateralInhibitionGate_3590592660177
// MI455X (gfx1250) — matrix operand order repaired and checked by reading, not compiled, not run on hardware
//
#include <hip/hip_runtime.h>

// ---------------------------------------------------------------------------
// LateralInhibitionGate for MI455X (gfx1250), wave32 + WMMA bf16 path.
//   B=4, T=512, H=1024, V=32000, K=64, temperature=1.
// Round 3: hoist the top-k threshold into one 32B LDS vector load per
// iteration (was 8 scattered ds_load_b32+wait pairs); vectorized converter.
// ---------------------------------------------------------------------------

#define B_   4
#define T_   512
#define H_   1024
#define V_   32000
#define BT_  (B_ * T_)      // 2048 tokens
#define K_   64             // top-k
#define SEG_ 10             // vocab segments per token-tile
#define SEGC_ (V_ / SEG_)   // 3200 columns per segment
#define TILES_ (BT_ / 16)   // 128 token tiles of 16
#define CHUNK_ 128          // columns per iteration (8 waves x 16 cols)

typedef __attribute__((ext_vector_type(16))) __bf16 v16bf;
typedef __attribute__((ext_vector_type(8)))  __bf16 v8bf;
typedef __attribute__((ext_vector_type(4)))  __bf16 v4bf;
typedef __attribute__((ext_vector_type(8)))  float  v8f;

// Build a 16x16x32 bf16 A-fragment for one lane from a row-major row pointer.
// Layout: lanes 0-15 hold K = {0..15}, lanes 16-31 hold K = {16..31} (matches B);
// rowp is pre-offset by the lane's K-base (0 or 16), so the two halves are the
// 16-byte chunks at +kc and +kc+8.
static __device__ inline v16bf make_afrag(const __bf16* rowp, int kc) {
  v8bf lo = *(const v8bf*)(rowp + kc);
  v8bf hi = *(const v8bf*)(rowp + kc + 8);
  v16bf a;
#pragma unroll
  for (int i = 0; i < 8; ++i) { a[i] = lo[i]; a[i + 8] = hi[i]; }
  return a;
}

// Async 16-byte global -> LDS copy (CDNA5, tracked by ASYNCcnt).
static __device__ inline void async_copy_b128(void* lds_ptr, const void* gptr) {
  const unsigned lds_off = (unsigned)(uintptr_t)lds_ptr;  // low 32b = LDS offset
  asm volatile("global_load_async_to_lds_b128 %0, %1, off"
               :: "v"(lds_off), "v"(gptr)
               : "memory");
}
static __device__ inline void wait_asynccnt0() {
  asm volatile("s_wait_asynccnt 0" ::: "memory");
}

// ---------------------------------------------------------------------------
// Kernel 1: f32 -> bf16 conversion of x and W_v into workspace (16B-in/8B-out
// vectors; pure bandwidth pass over ~140 MB).
// ---------------------------------------------------------------------------
__global__ __launch_bounds__(256)
void convert_bf16_kernel(const float* __restrict__ x,
                         const float* __restrict__ Wv,
                         __bf16* __restrict__ xb,
                         __bf16* __restrict__ wb) {
  const long nw4 = (long)V_ * H_ / 4;
  const long nx4 = (long)BT_ * H_ / 4;
  const long stride = (long)gridDim.x * blockDim.x;
  const float4* __restrict__ wv4 = (const float4*)Wv;
  const float4* __restrict__ x4  = (const float4*)x;
  v4bf* __restrict__ wb4 = (v4bf*)wb;
  v4bf* __restrict__ xb4 = (v4bf*)xb;
  for (long i = (long)blockIdx.x * blockDim.x + threadIdx.x; i < nw4; i += stride) {
    const float4 f = wv4[i];
    v4bf o; o[0] = (__bf16)f.x; o[1] = (__bf16)f.y; o[2] = (__bf16)f.z; o[3] = (__bf16)f.w;
    wb4[i] = o;
    if (i < nx4) {
      const float4 g = x4[i];
      v4bf p; p[0] = (__bf16)g.x; p[1] = (__bf16)g.y; p[2] = (__bf16)g.z; p[3] = (__bf16)g.w;
      xb4[i] = p;
    }
  }
}

// ---------------------------------------------------------------------------
// Kernel 2: fused acts = relu(x W_v^T + b_v) GEMM (bf16 WMMA) + per-token
// running top-64 over this block's vocab segment.
//   - x tile staged into LDS with global_load_async_to_lds_b128 (ASYNCcnt)
//   - selection: per-lane register threshold filter (one 32B LDS threshold
//     vector per iteration) -> LDS atomic queues, then a short serial
//     insertion pass per token.
// Grid: TILES_*SEG_ blocks, 256 threads (8 waves).
// ---------------------------------------------------------------------------
__global__ __launch_bounds__(256)
void gemm_topk_kernel(const __bf16* __restrict__ xb,
                      const __bf16* __restrict__ wb,
                      const float*  __restrict__ bv,
                      float* __restrict__ pvals,
                      int*   __restrict__ pidx) {
  __shared__ __bf16 sx[16][H_];          // 32 KB: 16-token x tile
  __shared__ float  topv[16][K_];        // 4 KB
  __shared__ int    topi[16][K_];        // 4 KB
  __shared__ float  qv[16][CHUNK_];      // 8 KB: candidate queues
  __shared__ int    qi[16][CHUNK_];      // 8 KB
  __shared__ int    qcnt[16];
  __shared__ alignas(32) float curmin[16];  // running top-list threshold

  const int tid  = threadIdx.x;
  const int wave = tid >> 5;
  const int lane = tid & 31;
  const int tile = blockIdx.x % TILES_;
  const int seg  = blockIdx.x / TILES_;

  // Stage the x tile (16 x 1024 bf16 = 2048 x 16B) straight into LDS via the
  // async global->LDS path; no VGPR staging.
  {
    const uint4* src = (const uint4*)(xb + (size_t)tile * 16 * H_);
    uint4* dst = (uint4*)&sx[0][0];
    for (int e = tid; e < 16 * H_ * 2 / 16; e += 256)
      async_copy_b128(dst + e, src + e);
    wait_asynccnt0();
  }
  if (tid < 16) { qcnt[tid] = 0; curmin[tid] = -1.0f; }  // relu acts >= 0 pass
  __syncthreads();

  // Per-token top list state (only meaningful for tid < 16).
  int cnt = 0; float mn = 0.f; int mnp = 0;

  const __bf16* Arow = &sx[lane & 15][(lane < 16) ? 0 : 16];
  const int mofs = (lane < 16) ? 0 : 8;

  for (int it = 0; it < SEGC_ / CHUNK_; ++it) {
    const int colbase = seg * SEGC_ + it * CHUNK_ + wave * 16;
    // B-fragment: lane holds column N=lane&15 (a row of W_v), 16 contiguous K.
    const __bf16* Brow =
        wb + (size_t)(colbase + (lane & 15)) * H_ + ((lane < 16) ? 0 : 16);

    // Threshold vector is loop-invariant within this iteration (only the
    // serial phase between barriers mutates it): one 32B LDS load.
    const v8f cm = *(const v8f*)(&curmin[mofs]);

    v8f acc = {};
#pragma unroll 4
    for (int kc = 0; kc < H_; kc += 32) {
      v16bf a = make_afrag(Arow, kc);
      v16bf b = *(const v16bf*)(Brow + kc);
      __builtin_prefetch(Brow + kc + 128, 0, 1);  // stream next B chunk
      acc = __builtin_amdgcn_wmma_f32_16x16x32_bf16(
          false, a, false, b, (short)0, acc, false, false);
    }

    // bias + relu + threshold filter straight from the accumulator fragment.
    // acc[r] is acts[token row r+mofs][column colbase + (lane&15)].
    const int col  = colbase + (lane & 15);
    const float bias = bv[col];
#pragma unroll
    for (int r = 0; r < 8; ++r) {
      float v = acc[r] + bias;
      v = v > 0.f ? v : 0.f;
      if (v > cm[r]) {                        // rare after warm-up
        const int trow = r + mofs;
        const int p = atomicAdd(&qcnt[trow], 1);
        qv[trow][p] = v;
        qi[trow][p] = col;
      }
    }
    __syncthreads();  // queues complete

    // Short serial insertion pass per token over the surviving candidates.
    if (tid < 16) {
      const int n = qcnt[tid];
      for (int c = 0; c < n; ++c) {
        const float v = qv[tid][c];
        const int  id = qi[tid][c];
        if (cnt < K_) {
          topv[tid][cnt] = v; topi[tid][cnt] = id;
          if (cnt == 0 || v < mn) { mn = v; mnp = cnt; }
          ++cnt;
        } else if (v > mn) {
          topv[tid][mnp] = v; topi[tid][mnp] = id;
          mn = topv[tid][0]; mnp = 0;
          for (int j = 1; j < K_; ++j) {
            const float tv = topv[tid][j];
            if (tv < mn) { mn = tv; mnp = j; }
          }
        }
      }
      qcnt[tid] = 0;
      curmin[tid] = (cnt < K_) ? -1.0f : mn;  // raise the filter threshold
    }
    __syncthreads();  // threshold/queue reset visible before next pushes
  }

  if (tid < 16) {
    const int token = tile * 16 + tid;
    const size_t base = ((size_t)token * SEG_ + seg) * K_;
    for (int j = 0; j < K_; ++j) {
      pvals[base + j] = (j < cnt) ? topv[tid][j] : -1e30f;
      pidx [base + j] = (j < cnt) ? topi[tid][j] : 0;
    }
  }
}

// ---------------------------------------------------------------------------
// Kernel 3: merge SEG_ partial top-64 lists into the exact per-token top-64
// via rank-by-count (ties broken by smaller index, like lax.top_k).
// ---------------------------------------------------------------------------
__global__ __launch_bounds__(64)
void topk_merge_kernel(const float* __restrict__ pvals,
                       const int*   __restrict__ pidx,
                       float* __restrict__ fvals,
                       int*   __restrict__ fidx) {
  __shared__ float cv[SEG_ * K_];
  __shared__ int   ci[SEG_ * K_];
  const int token = blockIdx.x;
  const int t = threadIdx.x;
  for (int e = t; e < SEG_ * K_; e += 64) {
    cv[e] = pvals[(size_t)token * SEG_ * K_ + e];
    ci[e] = pidx [(size_t)token * SEG_ * K_ + e];
  }
  __syncthreads();
  for (int e = t; e < SEG_ * K_; e += 64) {
    const float v = cv[e]; const int id = ci[e];
    int rank = 0;
    for (int j = 0; j < SEG_ * K_; ++j) {
      const float vj = cv[j]; const int ij = ci[j];
      rank += ((vj > v) || (vj == v && ij < id)) ? 1 : 0;
    }
    if (rank < K_) {
      fvals[(size_t)token * K_ + rank] = v;
      fidx [(size_t)token * K_ + rank] = id;
    }
  }
}

// ---------------------------------------------------------------------------
// Kernel 4: per-token Gram (WMMA), softmax, inhibition gating, W_f gather
// projection and residual add. One block per token, 256 threads.
// Dynamic LDS: 64x1024 bf16 normalized protos (128 KB) + 64x64 f32 sim + misc.
// ---------------------------------------------------------------------------
__global__ __launch_bounds__(256)
void token_finalize_kernel(const float* __restrict__ x,
                           const float* __restrict__ Wv,
                           const float* __restrict__ Wf,
                           const float* __restrict__ bf,
                           const float* __restrict__ alphap,
                           const float* __restrict__ fvals,
                           const int*   __restrict__ fidx,
                           float* __restrict__ out) {
  extern __shared__ char smem[];
  __bf16 (*P)[H_] = (__bf16(*)[H_])smem;                    // 64 x 1024 bf16
  float* sim  = (float*)(smem + (size_t)K_ * H_ * 2);       // 64 x 64
  float* vals = sim + K_ * K_;
  float* wsm  = vals + K_;
  float* res  = wsm + K_;
  int*   idx  = (int*)(res + K_);
  float* red  = (float*)(idx + K_);                          // max / sum

  const int token = blockIdx.x;
  const int tid = threadIdx.x;
  const int wave = tid >> 5, lane = tid & 31;
  const float alpha = alphap[0];

  if (tid < K_) {
    vals[tid] = fvals[(size_t)token * K_ + tid];
    idx[tid]  = fidx [(size_t)token * K_ + tid];
  }
  __syncthreads();

  // Gather + L2-normalize the 64 prototype rows into LDS as bf16.
  {
    const int row = tid >> 2, part = tid & 3;   // 4 threads per row, 256 elems each
    const float* src = Wv + (size_t)idx[row] * H_ + part * 256;
    float ss = 0.f;
    for (int k = 0; k < 256; ++k) { const float v = src[k]; ss += v * v; }
    ss += __shfl_xor(ss, 1, 32);
    ss += __shfl_xor(ss, 2, 32);
    float nrm = sqrtf(ss); nrm = fmaxf(nrm, 1e-12f);
    const float inv = 1.f / nrm;
    for (int k = 0; k < 256; ++k) P[row][part * 256 + k] = (__bf16)(src[k] * inv);
  }
  __syncthreads();

  // sim = Pn Pn^T : 4x4 grid of 16x16 WMMA tiles, 2 tiles per wave.
  for (int tI = 0; tI < 2; ++tI) {
    const int tileId = wave * 2 + tI;
    const int I = tileId >> 2, J = tileId & 3;
    const __bf16* Arow = &P[I * 16 + (lane & 15)][(lane < 16) ? 0 : 16];
    const __bf16* Brow = &P[J * 16 + (lane & 15)][(lane < 16) ? 0 : 16];
    v8f acc = {};
#pragma unroll 4
    for (int kc = 0; kc < H_; kc += 32) {
      v16bf a = make_afrag(Arow, kc);
      v16bf b = *(const v16bf*)(Brow + kc);
      acc = __builtin_amdgcn_wmma_f32_16x16x32_bf16(
          false, a, false, b, (short)0, acc, false, false);
    }
    const int mofs = (lane < 16) ? 0 : 8, n = lane & 15;
#pragma unroll
    for (int r = 0; r < 8; ++r) sim[(I * 16 + r + mofs) * K_ + J * 16 + n] = acc[r];
  }
  __syncthreads();

  // softmax over top-k values (temperature == 1)
  if (tid == 0) {
    float mx = vals[0];
    for (int j = 1; j < K_; ++j) mx = fmaxf(mx, vals[j]);
    red[0] = mx;
  }
  __syncthreads();
  if (tid < K_) wsm[tid] = __expf(vals[tid] - red[0]);
  __syncthreads();
  if (tid == 0) {
    float s = 0.f;
    for (int j = 0; j < K_; ++j) s += wsm[j];
    red[1] = s;
  }
  __syncthreads();
  if (tid < K_) wsm[tid] = wsm[tid] / red[1];
  __syncthreads();

  // inhibition + gate
  if (tid < K_) {
    float inh = 0.f;
    for (int l = 0; l < K_; ++l) {
      float s = sim[tid * K_ + l] - (l == tid ? 1.f : 0.f);
      inh += (s > 0.f ? s : 0.f) * wsm[l];
    }
    float r2 = vals[tid] * (1.f - alpha * inh);
    res[tid] = r2 > 0.f ? r2 : 0.f;
  }
  __syncthreads();

  // out[h] = x[h] + b_f[h] + sum_k res[k] * W_f[h, idx[k]]
  for (int hh = tid; hh < H_; hh += 256) {
    float acc2 = bf[hh];
    const float* Wfrow = Wf + (size_t)hh * V_;
    for (int k = 0; k < K_; ++k) acc2 += res[k] * Wfrow[idx[k]];
    out[(size_t)token * H_ + hh] = x[(size_t)token * H_ + hh] + acc2;
  }
}

// ---------------------------------------------------------------------------
// Host launcher.
// ---------------------------------------------------------------------------
extern "C" void kernel_launch(void* const* d_in, const int* in_sizes, int n_in,
                              void* d_out, int out_size, void* d_ws, size_t ws_size,
                              hipStream_t stream) {
  const float* x  = (const float*)d_in[0];
  const float* Wv = (const float*)d_in[1];
  const float* bv = (const float*)d_in[2];
  const float* Wf = (const float*)d_in[3];
  const float* bf = (const float*)d_in[4];
  const float* al = (const float*)d_in[5];
  float* out = (float*)d_out;

  // Workspace layout (~81 MB total):
  char* ws = (char*)d_ws;
  size_t o = 0;
  __bf16* xb = (__bf16*)(ws + o); o += (size_t)BT_ * H_ * 2;        // 4 MB
  __bf16* wb = (__bf16*)(ws + o); o += (size_t)V_ * H_ * 2;         // 65.5 MB
  float*  pv = (float*) (ws + o); o += (size_t)BT_ * SEG_ * K_ * 4; // 5.2 MB
  int*    pi = (int*)   (ws + o); o += (size_t)BT_ * SEG_ * K_ * 4; // 5.2 MB
  float*  fv = (float*) (ws + o); o += (size_t)BT_ * K_ * 4;        // 0.5 MB
  int*    fi = (int*)   (ws + o); o += (size_t)BT_ * K_ * 4;        // 0.5 MB

  convert_bf16_kernel<<<32768, 256, 0, stream>>>(x, Wv, xb, wb);
  gemm_topk_kernel<<<TILES_ * SEG_, 256, 0, stream>>>(xb, wb, bv, pv, pi);
  topk_merge_kernel<<<BT_, 64, 0, stream>>>(pv, pi, fv, fi);

  const size_t shmem = (size_t)K_ * H_ * 2       // protos bf16
                     + (size_t)K_ * K_ * 4       // sim
                     + 3 * K_ * 4                // vals, wsm, res
                     + K_ * 4                    // idx
                     + 16 * 4;                   // reductions
  token_finalize_kernel<<<BT_, 256, shmem, stream>>>(x, Wv, Wf, bf, al, fv, fi, out);
}
